// MultiHeadAttentionLayer_43843026158158
// MI455X (gfx1250) — matrix operand order repaired and checked by reading, not compiled, not run on hardware
//
#include <hip/hip_runtime.h>
#include <hip/hip_bf16.h>

typedef __bf16 bf16;
typedef bf16  v16bf __attribute__((ext_vector_type(16)));
typedef bf16  v8bf  __attribute__((ext_vector_type(8)));
typedef bf16  v4bf  __attribute__((ext_vector_type(4)));
typedef float v8f   __attribute__((ext_vector_type(8)));
typedef float v4f   __attribute__((ext_vector_type(4)));
typedef int   v4i   __attribute__((ext_vector_type(4)));

#define E_DIM 1024
#define NH    16
#define HD    64
#define BB    2
#define SS    4096
#define MM    (BB*SS)
#define SCALE_Q 0.125f  // 64^-0.5

#if defined(__has_builtin)
#  if __has_builtin(__builtin_amdgcn_global_load_async_to_lds_b128)
#    define HAS_ASYNC 1
#  endif
#endif
#ifndef HAS_ASYNC
#  define HAS_ASYNC 0
#endif

static __device__ __forceinline__ bf16 f2bf(float f) {
  unsigned u = __builtin_bit_cast(unsigned, f);
  unsigned r = u + 0x7FFFu + ((u >> 16) & 1u);   // round-to-nearest-even
  unsigned short h = (unsigned short)(r >> 16);
  return __builtin_bit_cast(bf16, h);
}

static __device__ __forceinline__ v8f wmma_bf16(v16bf a, v16bf b, v8f c) {
  return __builtin_amdgcn_wmma_f32_16x16x32_bf16(false, a, false, b, (short)0, c, false, false);
}

static __device__ __forceinline__ v16bf pack16(v8bf lo, v8bf hi) {
  v16bf r;
#pragma unroll
  for (int t = 0; t < 8; ++t) { r[t] = lo[t]; r[8+t] = hi[t]; }
  return r;
}

#if HAS_ASYNC
typedef __attribute__((address_space(1))) v4i gas_v4i;   // global (prints as __device__)
typedef __attribute__((address_space(3))) v4i las_v4i;   // LDS    (prints as __shared__)
static __device__ __forceinline__ void async_b128(const bf16* g, bf16* l) {
  __builtin_amdgcn_global_load_async_to_lds_b128(
      (gas_v4i*)g, (las_v4i*)l, 0, 0);
}
#endif

// ---------------- fp32 -> bf16 conversion (vectorized x4) ----------------
__global__ void cvt_f32_bf16(const float* __restrict__ in, bf16* __restrict__ out, int n4) {
  int i = blockIdx.x * blockDim.x + threadIdx.x;
  if (i < n4) {
    v4f x = ((const v4f*)in)[i];
    v4bf y;
#pragma unroll
    for (int j = 0; j < 4; ++j) y[j] = f2bf(x[j]);
    ((v4bf*)out)[i] = y;
  }
}

// ---------------- generic WMMA GEMM: C = A[M,K] * W[N,K]^T + bias --------
// MODE 0: f32 out [M, E]
// MODE 1: bf16 out per-head [b,h,s,d]                (Q and K), scaled
// MODE 2: bf16 out per-head transposed [b,h,d,s]     (V^T)
// Each wave computes a 64x32 output tile (4x2 accumulators). 4 waves/block.
template<int MODE>
__global__ void gemm_wmma(const bf16* __restrict__ A, const bf16* __restrict__ W,
                          const float* __restrict__ bias, void* __restrict__ outp,
                          float scale)
{
  const int lane = threadIdx.x & 31;
  const int widx = threadIdx.x >> 5;
  const int wt   = blockIdx.x * 4 + widx;
  const int tn   = wt & ((E_DIM/32) - 1);   // 32 column tiles of 32
  const int tm   = wt >> 5;                 // 128 row tiles of 64
  const int half = lane >> 4;
  const int l16  = lane & 15;
  const int row0 = tm * 64;
  const int col0 = tn * 32;

  v8f acc[4][2] = {};
  for (int kk = 0; kk < E_DIM; kk += 32) {
    v16bf a[4], bfr[2];
#pragma unroll
    for (int mi = 0; mi < 4; ++mi) {
      const bf16* ap = A + (size_t)(row0 + mi*16 + l16) * E_DIM + kk;
      a[mi] = pack16(*(const v8bf*)(ap + 16*half),
                     *(const v8bf*)(ap + 8 + 16*half));
    }
#pragma unroll
    for (int ni = 0; ni < 2; ++ni)
      bfr[ni] = *(const v16bf*)(W + (size_t)(col0 + ni*16 + l16) * E_DIM + kk + 16*half);
#pragma unroll
    for (int mi = 0; mi < 4; ++mi)
#pragma unroll
      for (int ni = 0; ni < 2; ++ni)
        acc[mi][ni] = wmma_bf16(a[mi], bfr[ni], acc[mi][ni]);
  }

#pragma unroll
  for (int mi = 0; mi < 4; ++mi)
#pragma unroll
    for (int ni = 0; ni < 2; ++ni) {
      const int n  = col0 + ni*16 + l16;
      const float bv = bias[n];
#pragma unroll
      for (int r = 0; r < 8; ++r) {
        const int m = row0 + mi*16 + r + 8*half;
        const float v = (acc[mi][ni][r] + bv) * scale;
        if (MODE == 0) {
          ((float*)outp)[(size_t)m * E_DIM + n] = v;
        } else {
          const int b_ = m >> 12, s_ = m & (SS - 1);
          const int h_ = n >> 6,  d_ = n & (HD - 1);
          bf16* o = (bf16*)outp;
          if (MODE == 1)
            o[(((size_t)b_*NH + h_)*SS + s_)*HD + d_] = f2bf(v);
          else
            o[(((size_t)b_*NH + h_)*HD + d_)*SS + s_] = f2bf(v);
        }
      }
    }
}

// ---------------- flash attention, fully transposed -----------------------
// S^T = K * Q^T and O^T = V^T * P^T: the WMMA C-layout N index (= lane%16)
// is the QUERY row everywhere -> per-lane scalar softmax stats & rescale.
// Each wave handles TWO 16-query tiles (32 queries): every K/V A-fragment
// feeds two wmma, halving LDS fragment traffic and barrier overhead per FLOP.
// The 4 waves of a block share one (b,h): K/V chunks are staged ONCE per
// block into LDS with GLOBAL_LOAD_ASYNC_TO_LDS (double-buffered, ASYNCcnt).
// __launch_bounds__(128,1): full VGPR budget, no spills (latency is hidden by
// the async double-buffer, not by occupancy).
// Q: [b,h,s,64] bf16 (pre-scaled), K: [b,h,s,64] bf16, Vt: [b,h,64,s] bf16
// Output: bf16 [b, s, E] (A-matrix for the output projection)
#define KROW 72   // 64 + 8 halves pad (144B row, 16B aligned, no 4-way conflicts)
#define VROW 40   // 32 + 8 halves pad (80B row)

#if HAS_ASYNC
static __device__ __forceinline__ void fill_kv(const bf16* Kb, const bf16* Vb, int kv,
                                               bf16 (*kb)[KROW], bf16 (*vb)[VROW], int tid)
{
#pragma unroll
  for (int i = 0; i < 2; ++i) {              // K chunk: 32 rows x 64 halves
    int idx = i*128 + tid;                   // 256 x 16B transfers
    int row = idx >> 3, c = idx & 7;
    async_b128(Kb + (size_t)(kv + row)*HD + c*8, &kb[row][c*8]);
  }
#pragma unroll
  for (int i = 0; i < 2; ++i) {              // V chunk: 64 rows x 32 halves
    int idx = i*128 + tid;
    int row = idx >> 2, c = idx & 3;
    async_b128(Vb + (size_t)row*SS + kv + c*8, &vb[row][c*8]);
  }
}
#endif

__global__ void __launch_bounds__(128, 1)
flash_attn(const bf16* __restrict__ Q, const bf16* __restrict__ K,
           const bf16* __restrict__ Vt, bf16* __restrict__ Out)
{
  __shared__ __align__(32) bf16 plds[4][2][16][32];  // per-wave P^T staging (2 q-tiles)
#if HAS_ASYNC
  __shared__ __align__(16) bf16 kbuf[2][32][KROW];
  __shared__ __align__(16) bf16 vbuf[2][64][VROW];
#endif
  const int tid  = threadIdx.x;
  const int lane = tid & 31;
  const int widx = tid >> 5;
  const int wt   = blockIdx.x * 4 + widx;
  const int qp   = wt & 127;      // S/32 q-pairs; block = 8 q-tiles, same (b,h)
  const int bh   = wt >> 7;       // 0..31
  const int half = lane >> 4;
  const int l16  = lane & 15;
  const int b_   = bh >> 4;
  const int h_   = bh & 15;
  const int q0   = qp * 32;

  const bf16* Qb = Q  + ((size_t)bh * SS + q0) * HD;
  const bf16* Kb = K  + (size_t)bh * SS * HD;
  const bf16* Vb = Vt + (size_t)bh * HD * SS;

  // Q^T as B-matrices (d x q): lane n=l16 is the query row; 2 q-tiles x 2 d-chunks
  v16bf bQ[2][2];
#pragma unroll
  for (int u = 0; u < 2; ++u)
#pragma unroll
    for (int f = 0; f < 2; ++f)
      bQ[u][f] = *(const v16bf*)(Qb + (size_t)(16*u + l16) * HD + 32*f + 16*half);

  v8f o[2][4] = {};                    // O^T accumulators per q-tile
  float mrow[2] = {-1e30f, -1e30f};    // per-lane scalar stats (query row l16)
  float lrow[2] = {0.0f, 0.0f};

#if HAS_ASYNC
  fill_kv(Kb, Vb, 0, kbuf[0], vbuf[0], tid);
#endif

  for (int kv = 0; kv < SS; kv += 32) {
#if HAS_ASYNC
    const int cur = (kv >> 5) & 1;
    if (kv + 32 < SS) {
      fill_kv(Kb, Vb, kv + 32, kbuf[cur ^ 1], vbuf[cur ^ 1], tid);
      asm volatile("s_wait_asynccnt 0x4" ::: "memory");  // current chunk landed
    } else {
      asm volatile("s_wait_asynccnt 0x0" ::: "memory");
    }
    __syncthreads();
#else
    if (kv + 32 < SS) {
      __builtin_prefetch(Kb + (size_t)(kv + 32 + l16) * HD, 0, 0);
      __builtin_prefetch(Vb + (size_t)l16 * SS + kv + 32, 0, 0);
    }
#endif

    // K A-fragments loaded once, reused by both q-tiles
    v16bf aK[2][2];
#pragma unroll
    for (int j = 0; j < 2; ++j) {
#if HAS_ASYNC
      const bf16* kp = &kbuf[cur][16*j + l16][0];
#else
      const bf16* kp = Kb + (size_t)(kv + 16*j + l16) * HD;
#endif
      aK[j][0] = pack16(*(const v8bf*)(kp +      16*half),
                        *(const v8bf*)(kp +  8 + 16*half));
      aK[j][1] = pack16(*(const v8bf*)(kp + 32 + 16*half),
                        *(const v8bf*)(kp + 40 + 16*half));
    }

    // S^T tiles: (kv x q); 8 wmma = 2 q-tiles x 2 kv-tiles x 2 d-chunks
    v8f sc[2][2];
#pragma unroll
    for (int u = 0; u < 2; ++u)
#pragma unroll
      for (int j = 0; j < 2; ++j) {
        v8f z = {};
        z = wmma_bf16(aK[j][0], bQ[u][0], z);
        z = wmma_bf16(aK[j][1], bQ[u][1], z);
        sc[u][j] = z;   // sc[u][j][r] = S[q0+16u+l16][kv+16j+r+8*half]
      }

    // online softmax per q-tile (per-lane scalar stats)
#pragma unroll
    for (int u = 0; u < 2; ++u) {
      float pm = fmaxf(sc[u][0][0], sc[u][1][0]);
#pragma unroll
      for (int r = 1; r < 8; ++r) pm = fmaxf(pm, fmaxf(sc[u][0][r], sc[u][1][r]));
      pm = fmaxf(pm, __shfl_xor(pm, 16, 32));

      const float mnew = fmaxf(mrow[u], pm);
      const float csc  = __expf(mrow[u] - mnew);
      mrow[u] = mnew;

      v8f p0, p1;
      float ps = 0.0f;
#pragma unroll
      for (int r = 0; r < 8; ++r) {
        p0[r] = __expf(sc[u][0][r] - mnew);
        p1[r] = __expf(sc[u][1][r] - mnew);
        ps += p0[r] + p1[r];
      }
      ps += __shfl_xor(ps, 16, 32);
      lrow[u] = lrow[u] * csc + ps;

#pragma unroll
      for (int t = 0; t < 4; ++t)
#pragma unroll
        for (int r = 0; r < 8; ++r) o[u][t][r] *= csc;   // scalar rescale

      // stage P^T as B-matrix: plds[q][kv], contiguous 8-element runs
      v8bf pj0, pj1;
#pragma unroll
      for (int r = 0; r < 8; ++r) { pj0[r] = f2bf(p0[r]); pj1[r] = f2bf(p1[r]); }
      *(v8bf*)&plds[widx][u][l16][     8*half] = pj0;
      *(v8bf*)&plds[widx][u][l16][16 + 8*half] = pj1;
    }
    asm volatile("s_wait_dscnt 0" ::: "memory");    // per-wave LDS in-order RAW
    v16bf pb[2];
#pragma unroll
    for (int u = 0; u < 2; ++u)
      pb[u] = *(const v16bf*)&plds[widx][u][l16][16*half];

    // O^T += V^T * P^T : each V A-fragment feeds both q-tiles
#pragma unroll
    for (int t = 0; t < 4; ++t) {
#if HAS_ASYNC
      const bf16* vp = &vbuf[cur][16*t + l16][0];
#else
      const bf16* vp = Vb + (size_t)(16*t + l16) * SS + kv;
#endif
      v16bf aV = pack16(*(const v8bf*)(vp +      16*half),
                        *(const v8bf*)(vp +  8 + 16*half));
#pragma unroll
      for (int u = 0; u < 2; ++u)
        o[u][t] = wmma_bf16(aV, pb[u], o[u][t]);
    }

#if HAS_ASYNC
    __syncthreads();   // all reads of buffer `cur` done before it is refilled
#endif
  }

  // normalize (per-lane scalar) and store 8 consecutive e-elements per lane
#pragma unroll
  for (int u = 0; u < 2; ++u) {
    const float inv = 1.0f / lrow[u];
    const size_t base = ((size_t)b_ * SS + q0 + 16*u + l16) * E_DIM + h_ * HD;
#pragma unroll
    for (int t = 0; t < 4; ++t) {
      v8bf ov;
#pragma unroll
      for (int r = 0; r < 8; ++r) ov[r] = f2bf(o[u][t][r] * inv);
      *(v8bf*)(Out + base + 16*t + 8*half) = ov;    // e = 16t + 8*half + r
    }
  }
}

// ---------------- host-side launch ----------------
extern "C" void kernel_launch(void* const* d_in, const int* in_sizes, int n_in,
                              void* d_out, int out_size, void* d_ws, size_t ws_size,
                              hipStream_t stream)
{
  (void)in_sizes; (void)n_in; (void)out_size; (void)ws_size;
  const float* hs = (const float*)d_in[0];
  const float* Wq = (const float*)d_in[1];
  const float* bq = (const float*)d_in[2];
  const float* Wk = (const float*)d_in[3];
  const float* bk = (const float*)d_in[4];
  const float* Wv = (const float*)d_in[5];
  const float* bv = (const float*)d_in[6];
  const float* Wo = (const float*)d_in[7];
  const float* bo = (const float*)d_in[8];

  char* ws = (char*)d_ws;
  bf16* Xb  = (bf16*)(ws);                          // 16 MB  X bf16 [M,E]
  bf16* Wqb = (bf16*)(ws + (16ull<<20));            //  2 MB each
  bf16* Wkb = (bf16*)(ws + (18ull<<20));
  bf16* Wvb = (bf16*)(ws + (20ull<<20));
  bf16* Wob = (bf16*)(ws + (22ull<<20));
  bf16* Qb  = (bf16*)(ws + (24ull<<20));            // 16 MB  [b,h,s,d]
  bf16* Kb  = (bf16*)(ws + (40ull<<20));            // 16 MB  [b,h,s,d]
  bf16* Vtb = (bf16*)(ws + (56ull<<20));            // 16 MB  [b,h,d,s]
  bf16* Ab  = (bf16*)(ws + (72ull<<20));            // 16 MB  attn out [b,s,E]

  const int nX4 = MM * E_DIM / 4;
  const int nW4 = E_DIM * E_DIM / 4;
  cvt_f32_bf16<<<(nX4 + 255)/256, 256, 0, stream>>>(hs, Xb, nX4);
  cvt_f32_bf16<<<(nW4 + 255)/256, 256, 0, stream>>>(Wq, Wqb, nW4);
  cvt_f32_bf16<<<(nW4 + 255)/256, 256, 0, stream>>>(Wk, Wkb, nW4);
  cvt_f32_bf16<<<(nW4 + 255)/256, 256, 0, stream>>>(Wv, Wvb, nW4);
  cvt_f32_bf16<<<(nW4 + 255)/256, 256, 0, stream>>>(Wo, Wob, nW4);

  const int GB = (MM/64) * (E_DIM/32) / 4;   // 1024 blocks, 4 waves each
  gemm_wmma<1><<<GB, 128, 0, stream>>>(Xb, Wqb, bq, Qb, SCALE_Q);
  gemm_wmma<1><<<GB, 128, 0, stream>>>(Xb, Wkb, bk, Kb, 1.0f);
  gemm_wmma<2><<<GB, 128, 0, stream>>>(Xb, Wvb, bv, Vtb, 1.0f);

  const int FB = (BB * NH * (SS/32)) / 4;    // 1024 blocks
  flash_attn<<<FB, 128, 0, stream>>>(Qb, Kb, Vtb, Ab);

  gemm_wmma<0><<<GB, 128, 0, stream>>>(Ab, Wob, bo, d_out, 1.0f);
}
